// HapticNet_18330920419616
// MI455X (gfx1250) — compile-verified, matrix operand order checked by reading, not run on hardware
//
#include <hip/hip_runtime.h>
#include <hip/hip_bf16.h>
#include <math.h>

// ---------------- model constants ----------------
#define CONV_CH   32
#define FRAG_DIM  8
#define N_P       6
#define N_T       12
#define PROJ      64
#define Bq        64
#define Tq        256
#define Wq        20
#define GATES     256          // 4*PROJ
#define CATK      72
#define CATKP     96           // padded to 3 x 32 for WMMA K-chunks
#define C2K       96           // conv2 GEMM K = 32ch * 3tap
#define C2KP      104          // im2col row stride: 16B-aligned + bank-staggered

typedef __attribute__((ext_vector_type(16))) _Float16 v16h;
typedef __attribute__((ext_vector_type(8)))  float    v8f;

// 16-bit A/B fragment K index for lane/element (ISA 7.12.2 layout)
__device__ __forceinline__ int kmap(int e, int hi8) {
    return (e & 7) + ((e >> 3) << 4) + hi8;   // hi8 = 8 * (lane >= 16)
}
__device__ __forceinline__ float sigm(float x) { return 1.f / (1.f + expf(-x)); }

// ---------------- kernel 0: weight prep (f32 -> f16, bias fuse) ----------------
__global__ void prep_kernel(const float* __restrict__ prw, const float* __restrict__ Wih,
                            const float* __restrict__ Whh, const float* __restrict__ bih,
                            const float* __restrict__ bhh,
                            const float* __restrict__ pw2, const float* __restrict__ tw2,
                            _Float16* __restrict__ prw16, _Float16* __restrict__ wih16,
                            _Float16* __restrict__ whh16, float* __restrict__ biasg,
                            _Float16* __restrict__ pw2f16, _Float16* __restrict__ tw2f16) {
    int tid = threadIdx.x;  // 256 threads, 1 block
    for (int i = tid; i < PROJ * CATKP; i += 256) {
        int o = i / CATKP, k = i % CATKP;
        prw16[i] = (_Float16)(k < CATK ? prw[o * CATK + k] : 0.f);
    }
    for (int i = tid; i < GATES * PROJ; i += 256) {
        wih16[i] = (_Float16)Wih[i];
        whh16[i] = (_Float16)Whh[i];
    }
    for (int i = tid; i < CONV_CH * C2K; i += 256) {
        pw2f16[i] = (_Float16)pw2[i];
        tw2f16[i] = (_Float16)tw2[i];
    }
    if (tid < GATES) biasg[tid] = bih[tid] + bhh[tid];
}

// ---------------- kernel 1: encoders -----------------------------------------
// conv1: VALU (K=18/36, minor FLOPs). conv2 (dominant, K=96): WMMA GEMM
//   out(32x16) = W2(32x96) @ im2col(96x16), bias in C, relu+mean in epilogue.
// im2col is stored TRANSPOSED (imT[col][k], stride 104 halfs): each lane's
// fragment halves are 16B-contiguous -> ds_load_b128, bank-conflict free.
__global__ void __launch_bounds__(64) encoder_kernel(
        const float* __restrict__ pressure, const float* __restrict__ torque,
        const float* __restrict__ fragility,
        const float* __restrict__ pw1, const float* __restrict__ pb1,
        const float* __restrict__ pb2,
        const float* __restrict__ tw1, const float* __restrict__ tb1,
        const float* __restrict__ tb2,
        const _Float16* __restrict__ pw2f16, const _Float16* __restrict__ tw2f16,
        const float* __restrict__ fw, const float* __restrict__ fb,
        _Float16* __restrict__ cat16) {
    const int w = blockIdx.x;          // window index = b*T + t
    const int b = w / Tq;
    const int tid = threadIdx.x;       // 0..63
    const int lane = tid & 31;
    const int wv   = tid >> 5;         // 0..1 -> conv2 M-tile
    const int mr   = lane & 15;
    const int hi8  = (lane >> 4) << 3;

    __shared__ float    xp[N_P][Wq];
    __shared__ float    xt[N_T][Wq];
    __shared__ _Float16 h1p[CONV_CH][18];
    __shared__ _Float16 h1t[CONV_CH][18];
    __shared__ __align__(16) _Float16 impT[16][C2KP];   // im2col^T, pressure
    __shared__ __align__(16) _Float16 imtT[16][C2KP];   // im2col^T, torque
    __shared__ float    red2[CONV_CH][16];
    __shared__ float    outrow[CATKP];

    const float* pin = pressure + (size_t)w * (Wq * N_P);
    for (int i = tid; i < Wq * N_P; i += 64) { xp[i % N_P][i / N_P] = pin[i]; }
    const float* tin = torque + (size_t)w * (Wq * N_T);
    for (int i = tid; i < Wq * N_T; i += 64) { xt[i % N_T][i / N_T] = tin[i]; }
    __syncthreads();

    // conv1 + relu (valid, K=3): 32 x 18 each, result to f16 (WMMA operand prec)
    for (int i = tid; i < CONV_CH * 18; i += 64) {
        int c = i / 18, l = i % 18;
        float s = pb1[c];
#pragma unroll
        for (int ic = 0; ic < N_P; ++ic)
#pragma unroll
            for (int k = 0; k < 3; ++k)
                s += pw1[(c * N_P + ic) * 3 + k] * xp[ic][l + k];
        h1p[c][l] = (_Float16)fmaxf(s, 0.f);
    }
    for (int i = tid; i < CONV_CH * 18; i += 64) {
        int c = i / 18, l = i % 18;
        float s = tb1[c];
#pragma unroll
        for (int ic = 0; ic < N_T; ++ic)
#pragma unroll
            for (int k = 0; k < 3; ++k)
                s += tw1[(c * N_T + ic) * 3 + k] * xt[ic][l + k];
        h1t[c][l] = (_Float16)fmaxf(s, 0.f);
    }
    __syncthreads();

    // transposed im2col: imT[l][kk], kk = ic*3 + tap  (l = output position 0..15)
    for (int i = tid; i < 16 * C2K; i += 64) {
        int l = i / C2K, kk = i % C2K;
        int ic = kk / 3, tap = kk % 3;
        impT[l][kk] = h1p[ic][l + tap];
        imtT[l][kk] = h1t[ic][l + tap];
    }
    __syncthreads();

    // conv2 via WMMA: D = W2 x im2col + bias
    v8f accp, acct;
#pragma unroll
    for (int v = 0; v < 8; ++v) {
        accp[v] = pb2[wv * 16 + v + hi8];
        acct[v] = tb2[wv * 16 + v + hi8];
    }
#pragma unroll
    for (int kc = 0; kc < C2K; kc += 32) {
        v16h a, bb;
#pragma unroll
        for (int e = 0; e < 16; ++e) {
            int k = kc + kmap(e, hi8);
            a[e]  = pw2f16[(size_t)(wv * 16 + mr) * C2K + k];
            bb[e] = impT[mr][k];                 // 16B-contiguous per 8 halves
        }
        accp = __builtin_amdgcn_wmma_f32_16x16x32_f16(false, a, false, bb,
                                                      (short)0, accp, false, false);
#pragma unroll
        for (int e = 0; e < 16; ++e) {
            int k = kc + kmap(e, hi8);
            a[e]  = tw2f16[(size_t)(wv * 16 + mr) * C2K + k];
            bb[e] = imtT[mr][k];
        }
        acct = __builtin_amdgcn_wmma_f32_16x16x32_f16(false, a, false, bb,
                                                      (short)0, acct, false, false);
    }

    // relu + mean pool (pressure), then reuse red2 for torque
#pragma unroll
    for (int v = 0; v < 8; ++v) red2[wv * 16 + v + hi8][mr] = fmaxf(accp[v], 0.f);
    __syncthreads();
    if (tid < CONV_CH) {
        float s = 0.f;
#pragma unroll
        for (int l = 0; l < 16; ++l) s += red2[tid][l];
        outrow[tid] = s * (1.f / 16.f);
    }
    __syncthreads();
#pragma unroll
    for (int v = 0; v < 8; ++v) red2[wv * 16 + v + hi8][mr] = fmaxf(acct[v], 0.f);
    __syncthreads();
    if (tid < CONV_CH) {
        float s = 0.f;
#pragma unroll
        for (int l = 0; l < 16; ++l) s += red2[tid][l];
        outrow[32 + tid] = s * (1.f / 16.f);
    }
    if (tid < FRAG_DIM) outrow[64 + tid] = fmaxf(fw[tid] * fragility[b] + fb[tid], 0.f);
    if (tid >= FRAG_DIM && tid < 32) outrow[64 + tid] = 0.f;   // pad K 72..95
    __syncthreads();
    for (int i = tid; i < CATKP; i += 64)
        cat16[(size_t)w * CATKP + i] = (_Float16)outrow[i];
}

// ---------------- kernel 2: WMMA projection GEMM (16384x96 @ 96x64) + relu --------
__global__ void __launch_bounds__(256) proj_kernel(
        const _Float16* __restrict__ cat16, const _Float16* __restrict__ prw16,
        const float* __restrict__ prb, _Float16* __restrict__ feat16) {
    const int tid  = threadIdx.x;
    const int lane = tid & 31;
    const int wv   = tid >> 5;               // 8 waves: 2 M-tiles x 4 N-tiles
    const int row0 = blockIdx.x * 32 + (wv >> 2) * 16;
    const int col0 = (wv & 3) * 16;
    const int mr   = lane & 15;
    const int hi8  = (lane >> 4) << 3;

    v8f acc = {};
#pragma unroll
    for (int kc = 0; kc < CATKP; kc += 32) {
        v16h a, bf;
#pragma unroll
        for (int e = 0; e < 16; ++e) {
            int k = kc + kmap(e, hi8);
            a[e]  = cat16[(size_t)(row0 + mr) * CATKP + k];      // A: row m, col k
            bf[e] = prw16[(size_t)(col0 + mr) * CATKP + k];      // B: col n, row k
        }
        acc = __builtin_amdgcn_wmma_f32_16x16x32_f16(false, a, false, bf,
                                                     (short)0, acc, false, false);
    }
    const int col  = col0 + mr;
    const float bv = prb[col];
#pragma unroll
    for (int v = 0; v < 8; ++v) {
        int row = row0 + v + hi8;
        feat16[(size_t)row * PROJ + col] = (_Float16)fmaxf(acc[v] + bv, 0.f);
    }
}

// ---------------- kernel 3: WMMA input-gate GEMM, TIME-MAJOR output ---------------
// xg[t][b][gate]: the serial LSTM kernel then reads one contiguous 16KB slice/step.
__global__ void __launch_bounds__(512) gates_kernel(
        const _Float16* __restrict__ feat16, const _Float16* __restrict__ wih16,
        const float* __restrict__ biasg, float* __restrict__ xg) {
    const int tid  = threadIdx.x;
    const int lane = tid & 31;
    const int wv   = tid >> 5;               // 16 waves = 16 N-tiles
    const int row0 = blockIdx.x * 16;
    const int col0 = wv * 16;
    const int mr   = lane & 15;
    const int hi8  = (lane >> 4) << 3;

    v8f acc;
    float bsc = biasg[col0 + mr];
#pragma unroll
    for (int v = 0; v < 8; ++v) acc[v] = bsc;     // bias folded into C
#pragma unroll
    for (int kc = 0; kc < PROJ; kc += 32) {
        v16h a, bf;
#pragma unroll
        for (int e = 0; e < 16; ++e) {
            int k = kc + kmap(e, hi8);
            a[e]  = feat16[(size_t)(row0 + mr) * PROJ + k];
            bf[e] = wih16[(size_t)(col0 + mr) * PROJ + k];
        }
        acc = __builtin_amdgcn_wmma_f32_16x16x32_f16(false, a, false, bf,
                                                     (short)0, acc, false, false);
    }
#pragma unroll
    for (int v = 0; v < 8; ++v) {
        int r = row0 + v + hi8;                   // window index = b*T + t
        int bb = r >> 8, t = r & (Tq - 1);
        xg[((size_t)t * Bq + bb) * GATES + col0 + mr] = acc[v];
    }
}

// ---------------- kernel 4: batch-split LSTM recurrence + head --------------------
// 4 blocks x 16 batch rows; per step each of 16 waves does one 16x16 tile of
// h(16x64) @ Whh^T(64x256) via two v_wmma_f32_16x16x32_f16, C preloaded with xg.
// Next step's xg slice is prefetched (global_prefetch_b8) during compute.
__global__ void __launch_bounds__(512) lstm_head_kernel(
        const float* __restrict__ xg, const _Float16* __restrict__ whh16,
        const float* __restrict__ hw, const float* __restrict__ hb,
        float* __restrict__ out, float* __restrict__ hT, float* __restrict__ cT) {
    const int tid  = threadIdx.x;
    const int lane = tid & 31;
    const int wv   = tid >> 5;               // 0..15
    const int b0   = blockIdx.x * 16;        // batch rows [b0, b0+16)
    const int col0 = wv * 16;
    const int mr   = lane & 15;
    const int hi8  = (lane >> 4) << 3;

    __shared__ __align__(16) _Float16 h16[16 * PROJ];   // h state, f16, WMMA A
    __shared__ float    hbuf[16 * PROJ];     // h state, f32, for head
    __shared__ float    gbuf[16 * GATES];    // per-step gate pre-activations

    for (int i = tid; i < 16 * PROJ; i += 512) { h16[i] = (_Float16)0.f; hbuf[i] = 0.f; }
    float cst0 = 0.f, cst1 = 0.f;            // c state: elems tid and tid+512 of 1024
    __syncthreads();

    // Whh B-fragments are loop-invariant: hoist into registers for all 256 steps.
    v16h bf0, bf1;
#pragma unroll
    for (int e = 0; e < 16; ++e) {
        int k = kmap(e, hi8);
        bf0[e] = whh16[(size_t)(col0 + mr) * PROJ + k];
        bf1[e] = whh16[(size_t)(col0 + mr) * PROJ + 32 + k];
    }

    for (int t = 0; t < Tq; ++t) {
        // phase 1: gbuf = xg[t,:,:] + h @ Whh^T   (C preloaded from time-major xg)
        v8f acc;
#pragma unroll
        for (int v = 0; v < 8; ++v)
            acc[v] = xg[((size_t)t * Bq + b0 + v + hi8) * GATES + col0 + mr];
        if (t + 1 < Tq) {        // hide next step's slice latency behind this step
#pragma unroll
            for (int v = 0; v < 8; ++v)
                __builtin_prefetch(&xg[((size_t)(t + 1) * Bq + b0 + v + hi8) * GATES
                                       + col0 + mr], 0, 1);
        }
        v16h a0, a1;
#pragma unroll
        for (int e = 0; e < 16; ++e) {
            int k = kmap(e, hi8);
            a0[e] = h16[mr * PROJ + k];
            a1[e] = h16[mr * PROJ + 32 + k];
        }
        acc = __builtin_amdgcn_wmma_f32_16x16x32_f16(false, a0, false, bf0,
                                                     (short)0, acc, false, false);
        acc = __builtin_amdgcn_wmma_f32_16x16x32_f16(false, a1, false, bf1,
                                                     (short)0, acc, false, false);
#pragma unroll
        for (int v = 0; v < 8; ++v)
            gbuf[(v + hi8) * GATES + col0 + mr] = acc[v];
        __syncthreads();

        // phase 2: gate nonlinearity + state update (1024 elems, 2/thread)
        {
            int e = tid, r = e >> 6, j = e & 63;
            float ig = sigm(gbuf[r * GATES + j]);
            float fg = sigm(gbuf[r * GATES + 64 + j]);
            float gg = tanhf(gbuf[r * GATES + 128 + j]);
            float og = sigm(gbuf[r * GATES + 192 + j]);
            cst0 = fg * cst0 + ig * gg;
            float hv = og * tanhf(cst0);
            hbuf[e] = hv; h16[e] = (_Float16)hv;
        }
        {
            int e = tid + 512, r = e >> 6, j = e & 63;
            float ig = sigm(gbuf[r * GATES + j]);
            float fg = sigm(gbuf[r * GATES + 64 + j]);
            float gg = tanhf(gbuf[r * GATES + 128 + j]);
            float og = sigm(gbuf[r * GATES + 192 + j]);
            cst1 = fg * cst1 + ig * gg;
            float hv = og * tanhf(cst1);
            hbuf[e] = hv; h16[e] = (_Float16)hv;
        }
        __syncthreads();

        // phase 3: head  out[b,t,:] = sigmoid(h @ hw^T + hb)
        if (tid < 64) {
            int r = tid >> 2, oo = tid & 3;
            float s = hb[oo];
#pragma unroll
            for (int j = 0; j < PROJ; ++j) s += hbuf[r * PROJ + j] * hw[oo * PROJ + j];
            out[((size_t)(b0 + r) * Tq + t) * 4 + oo] = sigm(s);
        }
        // next phase-1 only reads h16 / writes gbuf; phase-2 of t+1 is behind a
        // barrier, so no extra sync needed here.
    }

    for (int i = tid; i < 16 * PROJ; i += 512)
        hT[(size_t)(b0 + (i >> 6)) * PROJ + (i & 63)] = hbuf[i];
    {
        int e = tid;
        cT[(size_t)(b0 + (e >> 6)) * PROJ + (e & 63)] = cst0;
        e = tid + 512;
        cT[(size_t)(b0 + (e >> 6)) * PROJ + (e & 63)] = cst1;
    }
}

// ---------------- host launch ----------------
extern "C" void kernel_launch(void* const* d_in, const int* in_sizes, int n_in,
                              void* d_out, int out_size, void* d_ws, size_t ws_size,
                              hipStream_t stream) {
    const float* pressure  = (const float*)d_in[0];
    const float* torque    = (const float*)d_in[1];
    const float* fragility = (const float*)d_in[2];
    const float* pw1 = (const float*)d_in[3];  const float* pb1 = (const float*)d_in[4];
    const float* pw2 = (const float*)d_in[5];  const float* pb2 = (const float*)d_in[6];
    const float* tw1 = (const float*)d_in[7];  const float* tb1 = (const float*)d_in[8];
    const float* tw2 = (const float*)d_in[9];  const float* tb2 = (const float*)d_in[10];
    const float* fw  = (const float*)d_in[11]; const float* fb  = (const float*)d_in[12];
    const float* prw = (const float*)d_in[13]; const float* prb = (const float*)d_in[14];
    const float* Wih = (const float*)d_in[15]; const float* Whh = (const float*)d_in[16];
    const float* bih = (const float*)d_in[17]; const float* bhh = (const float*)d_in[18];
    const float* hw  = (const float*)d_in[19]; const float* hb  = (const float*)d_in[20];

    float* outp = (float*)d_out;
    float* hTp  = outp + (size_t)Bq * Tq * 4;    // 65536
    float* cTp  = hTp  + (size_t)Bq * PROJ;      // +4096

    // workspace carve-up (byte offsets, 256B aligned)
    char* ws = (char*)d_ws;
    size_t off = 0;
    _Float16* cat16  = (_Float16*)(ws + off); off += (size_t)Bq * Tq * CATKP * 2;  // 3.0 MB
    _Float16* feat16 = (_Float16*)(ws + off); off += (size_t)Bq * Tq * PROJ * 2;   // 2.0 MB
    off = (off + 255) & ~(size_t)255;
    float*    xg     = (float*)(ws + off);    off += (size_t)Bq * Tq * GATES * 4;  // 16 MB
    _Float16* prw16  = (_Float16*)(ws + off); off += (size_t)PROJ * CATKP * 2;
    _Float16* wih16  = (_Float16*)(ws + off); off += (size_t)GATES * PROJ * 2;
    _Float16* whh16  = (_Float16*)(ws + off); off += (size_t)GATES * PROJ * 2;
    _Float16* pw2f16 = (_Float16*)(ws + off); off += (size_t)CONV_CH * C2K * 2;
    _Float16* tw2f16 = (_Float16*)(ws + off); off += (size_t)CONV_CH * C2K * 2;
    off = (off + 255) & ~(size_t)255;
    float*    biasg  = (float*)(ws + off);    off += (size_t)GATES * 4;
    (void)ws_size; (void)in_sizes; (void)n_in; (void)out_size;

    prep_kernel<<<1, 256, 0, stream>>>(prw, Wih, Whh, bih, bhh, pw2, tw2,
                                       prw16, wih16, whh16, biasg, pw2f16, tw2f16);
    encoder_kernel<<<Bq * Tq, 64, 0, stream>>>(pressure, torque, fragility,
                                               pw1, pb1, pb2,
                                               tw1, tb1, tb2,
                                               pw2f16, tw2f16,
                                               fw, fb, cat16);
    proj_kernel<<<(Bq * Tq) / 32, 256, 0, stream>>>(cat16, prw16, prb, feat16);
    gates_kernel<<<(Bq * Tq) / 16, 512, 0, stream>>>(feat16, wih16, biasg, xg);
    lstm_head_kernel<<<Bq / 16, 512, 0, stream>>>(xg, whh16, hw, hb, outp, hTp, cTp);
}
